// LocalPointAttention_66451734004052
// MI455X (gfx1250) — matrix operand order repaired and checked by reading, not compiled, not run on hardware
//
#include <hip/hip_runtime.h>
#include <hip/hip_bf16.h>
#include <math.h>

#define B_   4
#define NQ_  4096
#define NS_  16384
#define KN_  16
#define P_   3
#define IN_  128
#define D_   256
#define NS1_ (NS_ + 1)
#define RADIUS_ 4.0f
#define TWO_PI_ 6.28318530717958647692f

typedef _Float16 half8  __attribute__((ext_vector_type(8)));
typedef _Float16 half16 __attribute__((ext_vector_type(16)));
typedef float    float8 __attribute__((ext_vector_type(8)));

// ---- WMMA helpers -------------------------------------------------------

__device__ __forceinline__ float8 wmma_f16(half16 a, half16 b, float8 c) {
  // D = A(16x32 f16) * B(32x16 f16) + C(16x16 f32)
  return __builtin_amdgcn_wmma_f32_16x16x32_f16(false, a, false, b, (short)0, c, false, false);
}

// A-fragment from row-major f16 LDS tile (stride in elements).
// 16-bit A 16x32 layout: lane<16: M=lane, K = {0..7, 16..23}; lane>=16: M=lane-16, K = {8..15, 24..31}
__device__ __forceinline__ half16 a_frag(const _Float16* base, int stride, int kc, int lane) {
  int row = lane & 15;
  int hf  = lane >> 4;
  const _Float16* p = base + row * stride + kc * 32 + hf * 8;
  half8 lo = *(const half8*)(p);        // K = kc*32 + 8*hf + {0..7}
  half8 hi = *(const half8*)(p + 16);   // K = kc*32 + 16 + 8*hf + {0..7}
  return __builtin_shufflevector(lo, hi, 0,1,2,3,4,5,6,7,8,9,10,11,12,13,14,15);
}

// ---- Weight packing into B-operand layout -------------------------------
// B 32x16 f16 layout: lane<16: N=lane, K = {0..7, 16..23}; lane>=16: N=lane-16, K = {8..15, 24..31}
// Packed as tiles, tile index = nt*nkc + kc, each tile = 32 lanes * 16 f16 contiguous.
__global__ void pack_b_kernel(const float* __restrict__ W, _Float16* __restrict__ dst, int kdim) {
  int nkc = kdim >> 5;
  int total = nkc * 16 * 32;           // (K/32) * (256/16) tiles * 32 lanes
  int gid = blockIdx.x * blockDim.x + threadIdx.x;
  if (gid >= total) return;
  int lane = gid & 31;
  int tile = gid >> 5;
  int kc = tile % nkc;
  int hf = lane >> 4;
  int n  = (tile / nkc) * 16 + (lane & 15);
  _Float16* outp = dst + (size_t)gid * 16;
  int kbase = kc * 32 + hf * 8;
  for (int p = 0; p < 16; ++p)
    outp[p] = (_Float16)W[(size_t)(kbase + (p < 8 ? p : p + 8)) * 256 + n];
}

// ---- Block-level 16x256 GEMM helper (256 threads = 8 waves, 2 col-tiles each)
// A-fragments hoisted into registers and shared across the wave's 2 col-tiles.
template <int KDIM>
__device__ __forceinline__ void block_gemm16(
    const _Float16* As, const _Float16* __restrict__ Bpk,
    const float* __restrict__ bias,
    float* addf32, float* storef32, _Float16* storef16, bool relu,
    int wave, int lane)
{
  constexpr int nkc = KDIM >> 5;
  int hf = lane >> 4;
  int nn = lane & 15;
  half16 af[nkc];
#pragma unroll
  for (int kc = 0; kc < nkc; ++kc) af[kc] = a_frag(As, KDIM, kc, lane);
#pragma unroll
  for (int ti = 0; ti < 2; ++ti) {
    int t = wave * 2 + ti;
    int n = t * 16 + nn;
    float bz = bias ? bias[n] : 0.0f;
    float8 c = { bz,bz,bz,bz,bz,bz,bz,bz };
    const _Float16* bp = Bpk + ((size_t)t * nkc * 32) * 16;
#pragma unroll
    for (int kc = 0; kc < nkc; ++kc) {
      half16 bb = *(const half16*)(bp + ((size_t)kc * 32 + lane) * 16);
      c = wmma_f16(af[kc], bb, c);
    }
#pragma unroll
    for (int r = 0; r < 8; ++r) {
      int m = r + 8 * hf;           // D layout: lane<16 -> M=r, lane>=16 -> M=8+r
      int o = m * 256 + n;
      if (addf32)   addf32[o] += c[r];
      if (storef32) storef32[o] = c[r];
      if (storef16) storef16[o] = (_Float16)(relu ? fmaxf(c[r], 0.0f) : c[r]);
    }
  }
}

// LayerNorm over 16 rows of 256, one wave handles 2 rows.
__device__ __forceinline__ void ln_rows(const float* Xs, const float* __restrict__ g,
                                        const float* __restrict__ bt, _Float16* Hs,
                                        int wave, int lane) {
  for (int rr = 0; rr < 2; ++rr) {
    int m = wave * 2 + rr;
    float s = 0.f;
    for (int j = lane; j < 256; j += 32) s += Xs[m*256+j];
    for (int o = 16; o > 0; o >>= 1) s += __shfl_xor(s, o, 32);
    float mean = s * (1.0f/256.0f);
    float v = 0.f;
    for (int j = lane; j < 256; j += 32) { float d = Xs[m*256+j] - mean; v += d*d; }
    for (int o = 16; o > 0; o >>= 1) v += __shfl_xor(v, o, 32);
    float inv = rsqrtf(v * (1.0f/256.0f) + 1e-5f);
    for (int j = lane; j < 256; j += 32)
      Hs[m*256+j] = (_Float16)((Xs[m*256+j] - mean) * inv * g[j] + bt[j]);
  }
}

// ---- Kernel 1: src_p = [src_feats; 0] @ W_proj + b_proj  (f32 out) -------
__global__ void __launch_bounds__(256) src_proj_kernel(
    const float* __restrict__ src_feats, const float* __restrict__ b_proj,
    const _Float16* __restrict__ Wpk, float* __restrict__ src_p)
{
  const int tiles_per_b = (NS1_ + 15) / 16;   // 1025
  int b  = blockIdx.x / tiles_per_b;
  int rt = blockIdx.x % tiles_per_b;
  int row0 = rt * 16;
  __shared__ __align__(32) _Float16 As[16 * IN_];
  for (int i = threadIdx.x; i < 16 * IN_; i += 256) {
    int m = i >> 7, k = i & (IN_ - 1);
    int r = row0 + m;
    float v = (r < NS_) ? src_feats[((size_t)b * NS_ + r) * IN_ + k] : 0.0f;
    As[i] = (_Float16)v;
  }
  __syncthreads();
  int wave = threadIdx.x >> 5, lane = threadIdx.x & 31;
  int hf = lane >> 4, nn = lane & 15;
  const bool full = (row0 + 16 <= NS1_);      // uniform: all but the last tile
  half16 af[IN_/32];
#pragma unroll
  for (int kc = 0; kc < IN_/32; ++kc) af[kc] = a_frag(As, IN_, kc, lane);
  for (int ti = 0; ti < 2; ++ti) {
    int t = wave * 2 + ti;
    int n = t * 16 + nn;
    float bz = b_proj[n];
    float8 c = { bz,bz,bz,bz,bz,bz,bz,bz };
#pragma unroll
    for (int kc = 0; kc < IN_/32; ++kc) {
      half16 bb = *(const half16*)(Wpk + ((size_t)(t*(IN_/32) + kc) * 32 + lane) * 16);
      c = wmma_f16(af[kc], bb, c);
    }
    float* dst = src_p + ((size_t)b * NS1_ + row0) * D_ + n;
    if (full) {
#pragma unroll
      for (int r = 0; r < 8; ++r)
        dst[(size_t)(r + 8*hf) * D_] = c[r];
    } else {
#pragma unroll
      for (int r = 0; r < 8; ++r) {
        int rr = row0 + r + 8*hf;
        if (rr < NS1_) dst[(size_t)(r + 8*hf) * D_] = c[r];
      }
    }
  }
}

// ---- Kernel 2: query pipeline (proj + pe0 + self-attn + LN2 + q-proj) ----
__global__ void __launch_bounds__(256) query_pre_kernel(
    const float* __restrict__ query_feats,
    const float* __restrict__ b_proj, const float* __restrict__ b_pos,
    const float* __restrict__ ln1_g, const float* __restrict__ ln1_b,
    const float* __restrict__ sa_bv, const float* __restrict__ sa_bo,
    const float* __restrict__ ln2_g, const float* __restrict__ ln2_b,
    const float* __restrict__ ca_bq,
    const _Float16* __restrict__ Wproj_pk, const _Float16* __restrict__ saWv_pk,
    const _Float16* __restrict__ saWo_pk, const _Float16* __restrict__ caWq_pk,
    float* __restrict__ x0_out, float* __restrict__ q_out)
{
  int row0 = blockIdx.x * 16;
  __shared__ __align__(32) _Float16 As[16 * IN_];
  __shared__ __align__(32) _Float16 Hs[16 * D_];
  __shared__ __align__(32) _Float16 Ts[16 * D_];
  __shared__ float Xs[16 * D_];
  int wave = threadIdx.x >> 5, lane = threadIdx.x & 31;

  // query_x = qf + pos_enc(0) ; pos_enc(0) = [0,0,0, cos(b_pos)]
  for (int i = threadIdx.x; i < 16 * D_; i += 256) {
    int n = i & (D_ - 1);
    Xs[i] = (n < P_) ? 0.0f : cosf(b_pos[n - P_]);
  }
  for (int i = threadIdx.x; i < 16 * IN_; i += 256) {
    int m = i >> 7, k = i & (IN_ - 1);
    As[i] = (_Float16)query_feats[((size_t)(row0 + m)) * IN_ + k];
  }
  __syncthreads();
  block_gemm16<IN_>(As, Wproj_pk, b_proj, Xs, nullptr, nullptr, false, wave, lane);
  __syncthreads();
  ln_rows(Xs, ln1_g, ln1_b, Hs, wave, lane);
  __syncthreads();
  block_gemm16<D_>(Hs, saWv_pk, sa_bv, nullptr, nullptr, Ts, false, wave, lane);
  __syncthreads();
  block_gemm16<D_>(Ts, saWo_pk, sa_bo, Xs, nullptr, nullptr, false, wave, lane);
  __syncthreads();
  ln_rows(Xs, ln2_g, ln2_b, Hs, wave, lane);
  for (int i = threadIdx.x; i < 16 * D_; i += 256)
    x0_out[(size_t)row0 * D_ + i] = Xs[i];
  __syncthreads();
  block_gemm16<D_>(Hs, caWq_pk, ca_bq, nullptr, q_out + (size_t)row0 * D_, nullptr, false, wave, lane);
}

// ---- Kernel 3: gather + pos-enc + cross-attn + FFN (dominant kernel) -----
// Each wave keeps TWO queries' A-fragments in registers so every packed
// B-fragment load (ca_Wk / ca_Wv) feeds two WMMAs -> half the L2 weight
// traffic, and lanes 0..15 / 16..31 handle query 0 / 1 in the scalar
// scores/softmax/ctx phases (no idle half-wave).
__global__ void __launch_bounds__(256) local_attn_kernel(
    const float* __restrict__ q_pts, const float* __restrict__ s_pts,
    const int*   __restrict__ neighb_inds,
    const float* __restrict__ W_pos, const float* __restrict__ b_pos,
    const float* __restrict__ ca_bk, const float* __restrict__ ca_bv,
    const float* __restrict__ ca_bo,
    const float* __restrict__ ln3_g, const float* __restrict__ ln3_b,
    const float* __restrict__ b1, const float* __restrict__ b2,
    const _Float16* __restrict__ caWk_pk, const _Float16* __restrict__ caWv_pk,
    const _Float16* __restrict__ caWo_pk, const _Float16* __restrict__ W1_pk,
    const _Float16* __restrict__ W2_pk,
    const float* __restrict__ src_p, const float* __restrict__ x0,
    const float* __restrict__ qv, float* __restrict__ out)
{
  int row0 = blockIdx.x * 16;
  int wave = threadIdx.x >> 5, lane = threadIdx.x & 31;

  __shared__ float Wpos_s[P_ * (D_ - P_)];
  __shared__ float bpos_s[D_ - P_];
  __shared__ float Xs[16 * D_];
  __shared__ __align__(32) _Float16 Ctx[16 * D_];   // ctx tile, reused as T1 for FFN
  __shared__ __align__(32) _Float16 Hs[16 * D_];
  __shared__ __align__(32) _Float16 NX[8][16 * D_]; // per-wave neighb_x tile (f16)
  __shared__ float Qs[8][2][D_];
  __shared__ float Tile[8][2][16 * 16];             // per-wave staged head tiles
  __shared__ float Sc[8][2][16 * 16];               // scores -> att
  __shared__ float Rel[8][2][16 * 4];               // rel xyz + dist2
  __shared__ int   Idx[8][2][16];

  for (int i = threadIdx.x; i < P_ * (D_ - P_); i += 256) Wpos_s[i] = W_pos[i];
  for (int i = threadIdx.x; i < (D_ - P_); i += 256) bpos_s[i] = b_pos[i];
  for (int i = threadIdx.x; i < 16 * D_; i += 256) Xs[i] = x0[(size_t)row0 * D_ + i];
  __syncthreads();

  const float invR = 1.0f / RADIUS_;
  int hf = lane >> 4, nn = lane & 15;

  // ---- neighbor geometry for both queries: lane = (sel, k) ----
  {
    int sel = lane >> 4;
    int k   = lane & 15;
    int qrow = row0 + wave * 2 + sel;
    int b = qrow / NQ_;
    int idx = neighb_inds[(size_t)qrow * KN_ + k] % NS1_;
    Idx[wave][sel][k] = idx;
    float d2 = 0.f;
    for (int p = 0; p < P_; ++p) {
      float sp = (idx < NS_) ? s_pts[((size_t)b * NS_ + idx) * P_ + p] : 1.0e6f;
      float r = sp - q_pts[(size_t)qrow * P_ + p];
      Rel[wave][sel][k*4+p] = r;
      d2 += r * r;
    }
    Rel[wave][sel][k*4+3] = d2;
    // prefetch this (sel,k) gathered row: 8 cachelines of 128B
    int b2i = qrow / NQ_;
    const char* rowp = (const char*)(src_p + ((size_t)b2i * NS1_ + idx) * D_);
#pragma unroll
    for (int i = 0; i < 8; ++i)
      __builtin_prefetch(rowp + i * 128, 0, 3);
  }
  // q vectors for both queries
#pragma unroll
  for (int sel = 0; sel < 2; ++sel) {
    int qrow = row0 + wave * 2 + sel;
    for (int j = lane; j < D_; j += 32)
      Qs[wave][sel][j] = qv[(size_t)qrow * D_ + j];
  }

  // ---- build neighb_x tiles and capture A-fragments for both queries ----
  half16 af0[8], af1[8];
#pragma unroll
  for (int sel = 0; sel < 2; ++sel) {
    int qrow = row0 + wave * 2 + sel;
    int b = qrow / NQ_;
    for (int k = 0; k < KN_; ++k) {
      int idx = Idx[wave][sel][k];
      const float* srow = src_p + ((size_t)b * NS1_ + idx) * D_;
      for (int j = lane; j < D_; j += 32) {
        float pe;
        if (j < P_) {
          pe = Rel[wave][sel][k*4+j] * invR;
        } else {
          float a = bpos_s[j - P_];
          for (int p = 0; p < P_; ++p)
            a += (TWO_PI_ * invR) * Rel[wave][sel][k*4+p] * Wpos_s[p * (D_ - P_) + (j - P_)];
          pe = cosf(a);
        }
        NX[wave][k * D_ + j] = (_Float16)(srow[j] + pe);
      }
    }
    if (sel == 0) {
#pragma unroll
      for (int kc = 0; kc < 8; ++kc) af0[kc] = a_frag(NX[wave], D_, kc, lane);
    } else {
#pragma unroll
      for (int kc = 0; kc < 8; ++kc) af1[kc] = a_frag(NX[wave], D_, kc, lane);
    }
  }

  // ---- K projection per head: one B load feeds both queries' WMMAs ----
  for (int h = 0; h < 16; ++h) {
    float bz = ca_bk[h*16 + nn];
    float8 c0 = { bz,bz,bz,bz,bz,bz,bz,bz };
    float8 c1 = c0;
    const _Float16* bp = caWk_pk + ((size_t)h * 8 * 32) * 16;
#pragma unroll
    for (int kc = 0; kc < 8; ++kc) {
      half16 bb = *(const half16*)(bp + ((size_t)kc * 32 + lane) * 16);
      c0 = wmma_f16(af0[kc], bb, c0);
      c1 = wmma_f16(af1[kc], bb, c1);
    }
#pragma unroll
    for (int r = 0; r < 8; ++r) {
      Tile[wave][0][(r + 8*hf) * 16 + nn] = c0[r];   // Tile[k][d]
      Tile[wave][1][(r + 8*hf) * 16 + nn] = c1[r];
    }
    {
      int sel = lane >> 4;
      int k   = lane & 15;
      float s = 0.f;
      for (int d = 0; d < 16; ++d)
        s += Qs[wave][sel][h*16 + d] * Tile[wave][sel][k*16 + d];
      bool masked = Rel[wave][sel][k*4+3] > RADIUS_ * RADIUS_;
      Sc[wave][sel][h*16 + k] = masked ? -1.0e9f : s * 0.25f;  // 1/sqrt(16)
    }
  }
  // ---- softmax over k: lane = (sel, head) ----
  {
    int sel = lane >> 4;
    int h   = lane & 15;
    float mx = -1.0e30f;
    for (int k = 0; k < KN_; ++k) mx = fmaxf(mx, Sc[wave][sel][h*16+k]);
    float sum = 0.f;
    for (int k = 0; k < KN_; ++k) {
      float e = __expf(Sc[wave][sel][h*16+k] - mx);
      Sc[wave][sel][h*16+k] = e;
      sum += e;
    }
    float inv = 1.0f / sum;
    for (int k = 0; k < KN_; ++k) Sc[wave][sel][h*16+k] *= inv;
  }
  // ---- V projection per head + ctx ----
  for (int h = 0; h < 16; ++h) {
    float bz = ca_bv[h*16 + nn];
    float8 c0 = { bz,bz,bz,bz,bz,bz,bz,bz };
    float8 c1 = c0;
    const _Float16* bp = caWv_pk + ((size_t)h * 8 * 32) * 16;
#pragma unroll
    for (int kc = 0; kc < 8; ++kc) {
      half16 bb = *(const half16*)(bp + ((size_t)kc * 32 + lane) * 16);
      c0 = wmma_f16(af0[kc], bb, c0);
      c1 = wmma_f16(af1[kc], bb, c1);
    }
#pragma unroll
    for (int r = 0; r < 8; ++r) {
      Tile[wave][0][(r + 8*hf) * 16 + nn] = c0[r];
      Tile[wave][1][(r + 8*hf) * 16 + nn] = c1[r];
    }
    {
      int sel = lane >> 4;
      int d   = lane & 15;
      float acc = 0.f;
      for (int k = 0; k < KN_; ++k)
        acc += Sc[wave][sel][h*16+k] * Tile[wave][sel][k*16 + d];
      Ctx[(wave*2 + sel) * D_ + h*16 + d] = (_Float16)acc;
    }
  }
  __syncthreads();
  // x += ctx @ Wo + bo
  block_gemm16<D_>(Ctx, caWo_pk, ca_bo, Xs, nullptr, nullptr, false, wave, lane);
  __syncthreads();
  ln_rows(Xs, ln3_g, ln3_b, Hs, wave, lane);
  __syncthreads();
  // T1 = relu(h @ W1 + b1)  (reuse Ctx)
  block_gemm16<D_>(Hs, W1_pk, b1, nullptr, nullptr, Ctx, true, wave, lane);
  __syncthreads();
  // x += T1 @ W2 + b2
  block_gemm16<D_>(Ctx, W2_pk, b2, Xs, nullptr, nullptr, false, wave, lane);
  __syncthreads();
  for (int i = threadIdx.x; i < 16 * D_; i += 256) {
    float v = Xs[i];
    out[(size_t)row0 * D_ + i] = (v == v) ? v : 0.0f;   // nan_to_num
  }
}

// ---- Host launcher -------------------------------------------------------
extern "C" void kernel_launch(void* const* d_in, const int* in_sizes, int n_in,
                              void* d_out, int out_size, void* d_ws, size_t ws_size,
                              hipStream_t stream)
{
  const float* q_pts       = (const float*)d_in[0];
  const float* s_pts       = (const float*)d_in[1];
  const float* src_feats   = (const float*)d_in[2];
  const float* query_feats = (const float*)d_in[3];
  const float* W_proj      = (const float*)d_in[4];
  const float* b_proj      = (const float*)d_in[5];
  const float* W_pos       = (const float*)d_in[6];
  const float* b_pos       = (const float*)d_in[7];
  const float* ln1_g       = (const float*)d_in[8];
  const float* ln1_b       = (const float*)d_in[9];
  const float* sa_Wv       = (const float*)d_in[10];
  const float* sa_bv       = (const float*)d_in[11];
  const float* sa_Wo       = (const float*)d_in[12];
  const float* sa_bo       = (const float*)d_in[13];
  const float* ln2_g       = (const float*)d_in[14];
  const float* ln2_b       = (const float*)d_in[15];
  const float* ca_Wq       = (const float*)d_in[16];
  const float* ca_bq       = (const float*)d_in[17];
  const float* ca_Wk       = (const float*)d_in[18];
  const float* ca_bk       = (const float*)d_in[19];
  const float* ca_Wv       = (const float*)d_in[20];
  const float* ca_bv       = (const float*)d_in[21];
  const float* ca_Wo       = (const float*)d_in[22];
  const float* ca_bo       = (const float*)d_in[23];
  const float* ln3_g       = (const float*)d_in[24];
  const float* ln3_b       = (const float*)d_in[25];
  const float* W1          = (const float*)d_in[26];
  const float* b1          = (const float*)d_in[27];
  const float* W2          = (const float*)d_in[28];
  const float* b2          = (const float*)d_in[29];
  const int*   neighb_inds = (const int*)d_in[30];
  float* outp = (float*)d_out;

  char* ws = (char*)d_ws;
  size_t off = 0;
  auto carve = [&](size_t bytes) -> void* {
    void* p = ws + off;
    off += (bytes + 255) & ~(size_t)255;
    return p;
  };
  float* src_p = (float*)carve(sizeof(float) * (size_t)B_ * NS1_ * D_);
  float* x0    = (float*)carve(sizeof(float) * (size_t)B_ * NQ_ * D_);
  float* qvv   = (float*)carve(sizeof(float) * (size_t)B_ * NQ_ * D_);

  const float* wsrc[9] = { W_proj, sa_Wv, sa_Wo, ca_Wq, ca_Wk, ca_Wv, ca_Wo, W1, W2 };
  int kd[9] = { IN_, D_, D_, D_, D_, D_, D_, D_, D_ };
  _Float16* pk[9];
  for (int i = 0; i < 9; ++i) {
    size_t nel = (size_t)(kd[i] / 32) * 16 * 32 * 16;   // tiles * 32 lanes * 16 f16
    pk[i] = (_Float16*)carve(nel * sizeof(_Float16));
  }
  for (int i = 0; i < 9; ++i) {
    int total = (kd[i] / 32) * 16 * 32;
    pack_b_kernel<<<dim3((total + 255) / 256), dim3(256), 0, stream>>>(wsrc[i], pk[i], kd[i]);
  }
  src_proj_kernel<<<dim3(B_ * ((NS1_ + 15) / 16)), dim3(256), 0, stream>>>(
      src_feats, b_proj, pk[0], src_p);
  query_pre_kernel<<<dim3((B_ * NQ_) / 16), dim3(256), 0, stream>>>(
      query_feats, b_proj, b_pos, ln1_g, ln1_b, sa_bv, sa_bo, ln2_g, ln2_b, ca_bq,
      pk[0], pk[1], pk[2], pk[3], x0, qvv);
  local_attn_kernel<<<dim3((B_ * NQ_) / 16), dim3(256), 0, stream>>>(
      q_pts, s_pts, neighb_inds, W_pos, b_pos, ca_bk, ca_bv, ca_bo,
      ln3_g, ln3_b, b1, b2, pk[4], pk[5], pk[6], pk[7], pk[8],
      src_p, x0, qvv, outp);

  (void)in_sizes; (void)n_in; (void)out_size; (void)ws_size;
}
